// UnaGenModel_34849364640177
// MI455X (gfx1250) — matrix operand order repaired and checked by reading, not compiled, not run on hardware
//
#include <hip/hip_runtime.h>
#include <cstdint>
#include <cstddef>

typedef __attribute__((ext_vector_type(16))) _Float16 v16h;
typedef __attribute__((ext_vector_type(8)))  _Float16 v8h;
typedef __attribute__((ext_vector_type(8)))  float    v8f;

// ---- WMMA f16 A-fragment K-offset (contiguous order: k = 16*half + e, matches B) ----
__device__ __forceinline__ int a_koff(int e, int half) {
  return 16 * half + e;
}

// ---- ordered-uint mapping for float atomic min/max ----
__device__ __forceinline__ unsigned f2ord(float f) {
  unsigned u = __float_as_uint(f);
  return (u & 0x80000000u) ? ~u : (u | 0x80000000u);
}
__device__ __forceinline__ float ord2f(unsigned u) {
  unsigned v = (u & 0x80000000u) ? (u ^ 0x80000000u) : ~u;
  return __uint_as_float(v);
}

// =====================================================================
// Implicit-GEMM conv (ks=3 pad=1 / ks=1 pad=0), fused bias + optional ReLU.
// Block = 128 threads (4 waves). Block tile: 64 out-channels x 256 pixels.
// Per K-chunk: 64x32 weight tile staged once in LDS (f16); each wave builds
// its 4 im2col B fragments ONCE and reuses them across 4 co-tiles
// -> 16 v_wmma_f32_16x16x32_f16 per 64 gathers (4x the density of a
// one-co-tile scheme, and no cross-block regather over Cout).
// =====================================================================
__global__ void k_conv_wmma(const float* __restrict__ X, const float* __restrict__ Wt,
                            const float* __restrict__ Bias, float* __restrict__ Y,
                            int Cin, int Cout, int H, int Wd, int ks, int pad, int relu) {
  __shared__ _Float16 sA[64][32];            // [co][k] f16 weight tile
  const int tid = threadIdx.x;
  const int lane = tid & 31, wave = tid >> 5;
  const int m = lane & 15, half = lane >> 4;
  const int P = H * Wd;
  const int coBase = blockIdx.y * 64;
  const int b = blockIdx.z;
  const int pBase = blockIdx.x * 256 + wave * 64;
  const int kk2 = ks * ks;
  const int K = Cin * kk2;
  int nCT = (Cout - coBase) >> 4;            // valid 16-wide co tiles (uniform)
  if (nCT > 4) nCT = 4;

  int pp[4], hh[4], ww[4];
#pragma unroll
  for (int pt = 0; pt < 4; ++pt) {
    int p = pBase + pt * 16 + m;
    pp[pt] = p;
    int pc = p < P ? p : P - 1;
    hh[pt] = pc / Wd; ww[pt] = pc % Wd;
  }

  v8f acc[4][4];                             // [co-tile][pixel-tile]
#pragma unroll
  for (int i = 0; i < 4; ++i)
#pragma unroll
    for (int j = 0; j < 4; ++j)
#pragma unroll
      for (int q = 0; q < 8; ++q) acc[i][j][q] = 0.0f;

  const float* Xb = X + (size_t)b * Cin * P;

  int kb = 0;
  for (; kb + 32 <= K; kb += 32) {           // full chunks: no K guards
    __syncthreads();
    // cooperative weight-tile load: 2048 elems, 16 contiguous-K per thread
#pragma unroll
    for (int j = 0; j < 16; ++j) {
      int idx = tid * 16 + j;
      int kr = idx & 31, r = idx >> 5;
      int co = coBase + r;
      int cc = co < Cout ? co : Cout - 1;
      sA[r][kr] = (_Float16)Wt[(size_t)cc * K + kb + kr];
    }
    __syncthreads();

    // B fragments built ONCE per chunk: incremental (kx,ky,ci) decode
    v16h bf[4];
    int kk0 = kb + 16 * half;
    int ci = kk0 / kk2, rr = kk0 % kk2;
    int ky = rr / ks, kx = rr % ks;
#pragma unroll
    for (int e = 0; e < 16; ++e) {
      const float* Xc = Xb + (size_t)ci * P;
#pragma unroll
      for (int pt = 0; pt < 4; ++pt) {
        int ih = hh[pt] + ky - pad, iw = ww[pt] + kx - pad;
        float xv = 0.0f;
        if (pp[pt] < P && ih >= 0 && ih < H && iw >= 0 && iw < Wd)
          xv = Xc[ih * Wd + iw];
        bf[pt][e] = (_Float16)xv;
      }
      kx++;
      if (kx == ks) { kx = 0; ky++; if (ky == ks) { ky = 0; ci++; } }
    }

    // reuse B fragments across 4 co-tiles (wave-uniform skip keeps EXEC full)
#pragma unroll
    for (int ct = 0; ct < 4; ++ct) {
      if (ct < nCT) {
        v8h alo = *(const v8h*)&sA[ct * 16 + m][16 * half];
        v8h ahi = *(const v8h*)&sA[ct * 16 + m][16 * half + 8];
        v16h afrag;
#pragma unroll
        for (int e = 0; e < 8; ++e) { afrag[e] = alo[e]; afrag[8 + e] = ahi[e]; }
#pragma unroll
        for (int pt = 0; pt < 4; ++pt)
          acc[ct][pt] = __builtin_amdgcn_wmma_f32_16x16x32_f16(
              false, afrag, false, bf[pt], (short)0, acc[ct][pt], false, false);
      }
    }
  }

  if (kb < K) {                              // guarded tail (e.g. Cin=3 -> K=27)
    v16h bf[4];
#pragma unroll
    for (int e = 0; e < 16; ++e) {
      int kk = kb + 16 * half + e;
      int vk = kk < K;
      int kc = vk ? kk : 0;
      int ci = kc / kk2, r2 = kc % kk2;
      int ky = r2 / ks, kx = r2 % ks;
      const float* Xc = Xb + (size_t)ci * P;
#pragma unroll
      for (int pt = 0; pt < 4; ++pt) {
        int ih = hh[pt] + ky - pad, iw = ww[pt] + kx - pad;
        float xv = 0.0f;
        if (vk && pp[pt] < P && ih >= 0 && ih < H && iw >= 0 && iw < Wd)
          xv = Xc[ih * Wd + iw];
        bf[pt][e] = (_Float16)xv;
      }
    }
#pragma unroll
    for (int ct = 0; ct < 4; ++ct) {
      if (ct < nCT) {
        int co = coBase + ct * 16 + m;
        v16h afrag;
#pragma unroll
        for (int e = 0; e < 16; ++e) {
          int ka = kb + a_koff(e, half);
          float wv = (ka < K) ? Wt[(size_t)co * K + ka] : 0.0f;
          afrag[e] = (_Float16)wv;
        }
#pragma unroll
        for (int pt = 0; pt < 4; ++pt)
          acc[ct][pt] = __builtin_amdgcn_wmma_f32_16x16x32_f16(
              false, afrag, false, bf[pt], (short)0, acc[ct][pt], false, false);
      }
    }
  }

  float* Yb = Y + (size_t)b * Cout * P;
#pragma unroll
  for (int ct = 0; ct < 4; ++ct) {
    if (ct < nCT) {
#pragma unroll
      for (int pt = 0; pt < 4; ++pt) {
#pragma unroll
        for (int r = 0; r < 8; ++r) {
          int co = coBase + ct * 16 + r + 8 * half;  // C/D layout: VGPR r -> M=r / r+8
          int p = pp[pt];
          if (p < P) {
            float v = acc[ct][pt][r] + Bias[co];
            if (relu) v = fmaxf(v, 0.0f);
            Yb[(size_t)co * P + p] = v;
          }
        }
      }
    }
  }
}

// =====================================================================
// GEMM: C[M,N] = act(A[M,K] @ B[K,N] + bias[N]); f32 in, f16 WMMA, f32 acc.
// Block = 128 threads (4 waves); macro-tile 64 rows x 64 cols. A and B
// K-chunks staged in LDS as f16 ([row][k] / [col][k]) so fragments are
// contiguous wide DS loads. Each wave: 16 rows x 4 col-tiles -> 4 wmma/chunk.
// =====================================================================
__global__ void k_gemm_wmma(const float* __restrict__ A, const float* __restrict__ Bw,
                            const float* __restrict__ bias, float* __restrict__ C,
                            int M, int N, int K, int relu) {
  __shared__ _Float16 sA[64][32];             // [row][k]
  __shared__ _Float16 sB[64][32];             // [col][k]
  const int tid = threadIdx.x;
  const int lane = tid & 31, wave = tid >> 5;
  const int m = lane & 15, half = lane >> 4;
  const int rowBase = blockIdx.x * 64;
  const int colBase = blockIdx.y * 64;
  const int lr = wave * 16 + m;               // local A row for this lane

  v8f acc[4];
#pragma unroll
  for (int i = 0; i < 4; ++i)
#pragma unroll
    for (int j = 0; j < 8; ++j) acc[i][j] = 0.0f;

  int kb = 0;
  for (; kb + 32 <= K; kb += 32) {            // full chunks
    __syncthreads();
    // A tile: 16 contiguous K-elements per thread (merges to b128 loads)
#pragma unroll
    for (int j = 0; j < 16; ++j) {
      int idx = tid * 16 + j;
      int kr = idx & 31, r = idx >> 5;
      int row = rowBase + r;
      int rc = row < M ? row : M - 1;
      sA[r][kr] = (_Float16)A[(size_t)rc * K + kb + kr];
    }
    // B tile: coalesced across threads along N
    for (int i = tid; i < 2048; i += 128) {
      int c = i & 63, kr = i >> 6;
      int col = colBase + c;
      float bv = (col < N) ? Bw[(size_t)(kb + kr) * N + col] : 0.0f;
      sB[c][kr] = (_Float16)bv;
    }
    __syncthreads();

    v8h alo = *(const v8h*)&sA[lr][16 * half];
    v8h ahi = *(const v8h*)&sA[lr][16 * half + 8];
    v16h afrag;
#pragma unroll
    for (int e = 0; e < 8; ++e) { afrag[e] = alo[e]; afrag[8 + e] = ahi[e]; }

#pragma unroll
    for (int ct = 0; ct < 4; ++ct) {
      if (colBase + ct * 16 < N) {            // wave-uniform tile skip
        v16h bfrag = *(const v16h*)&sB[ct * 16 + m][16 * half];
        acc[ct] = __builtin_amdgcn_wmma_f32_16x16x32_f16(false, afrag, false, bfrag,
                                                         (short)0, acc[ct], false, false);
      }
    }
  }

  if (kb < K) {                               // guarded tail chunk
    int row = rowBase + lr;
    int rc = row < M ? row : M - 1;
    v16h afrag;
#pragma unroll
    for (int e = 0; e < 16; ++e) {
      int ka = kb + a_koff(e, half);
      float av = (ka < K && row < M) ? A[(size_t)rc * K + ka] : 0.0f;
      afrag[e] = (_Float16)av;
    }
#pragma unroll
    for (int ct = 0; ct < 4; ++ct) {
      if (colBase + ct * 16 < N) {
        int col = colBase + ct * 16 + m;
        v16h bfrag;
#pragma unroll
        for (int e = 0; e < 16; ++e) {
          int kk = kb + 16 * half + e;
          float bv = (kk < K && col < N) ? Bw[(size_t)kk * N + col] : 0.0f;
          bfrag[e] = (_Float16)bv;
        }
        acc[ct] = __builtin_amdgcn_wmma_f32_16x16x32_f16(false, afrag, false, bfrag,
                                                         (short)0, acc[ct], false, false);
      }
    }
  }

#pragma unroll
  for (int ct = 0; ct < 4; ++ct) {
#pragma unroll
    for (int r = 0; r < 8; ++r) {
      int row = rowBase + wave * 16 + r + 8 * half;
      int col = colBase + ct * 16 + m;
      if (row < M && col < N) {
        float v = acc[ct][r] + bias[col];
        if (relu) v = fmaxf(v, 0.0f);
        C[(size_t)row * N + col] = v;
      }
    }
  }
}

// ===================== bandwidth glue kernels ========================
__global__ void k_pool(const float* __restrict__ X, float* __restrict__ Y,
                       int C, int H, int W, int Bn) {
  int Ho = H / 2, Wo = W / 2;
  long long total = (long long)Bn * C * Ho * Wo;
  long long t = (long long)blockIdx.x * blockDim.x + threadIdx.x;
  if (t >= total) return;
  int wo = (int)(t % Wo); long long r = t / Wo;
  int ho = (int)(r % Ho); r /= Ho;
  int c = (int)(r % C);   int b = (int)(r / C);
  const float* xb = X + ((size_t)(b * C + c) * H) * W;
  float a0 = xb[(2 * ho) * W + 2 * wo],     a1 = xb[(2 * ho) * W + 2 * wo + 1];
  float a2 = xb[(2 * ho + 1) * W + 2 * wo], a3 = xb[(2 * ho + 1) * W + 2 * wo + 1];
  Y[t] = fmaxf(fmaxf(a0, a1), fmaxf(a2, a3));
}

__global__ void k_tconv(const float* __restrict__ X, const float* __restrict__ Wt,
                        const float* __restrict__ Bias, float* __restrict__ Y,
                        int Cin, int Cout, int Hin, int Win, int Ctot, int Bn) {
  int Ho = Hin * 2, Wo = Win * 2;
  long long total = (long long)Bn * Cout * Ho * Wo;
  long long t = (long long)blockIdx.x * blockDim.x + threadIdx.x;
  if (t >= total) return;
  int wo = (int)(t % Wo); long long r = t / Wo;
  int ho = (int)(r % Ho); r /= Ho;
  int co = (int)(r % Cout); int b = (int)(r / Cout);
  int h = ho >> 1, dy = ho & 1, w = wo >> 1, dx = wo & 1;
  float acc = Bias[co];
  for (int ci = 0; ci < Cin; ++ci)
    acc += X[(((size_t)b * Cin + ci) * Hin + h) * Win + w] *
           Wt[(((size_t)ci * Cout + co) * 2 + dy) * 2 + dx];
  Y[(((size_t)b * Ctot + co) * Ho + ho) * Wo + wo] = acc;
}

__global__ void k_copych(const float* __restrict__ X, float* __restrict__ Y,
                         int C, int Ctot, int chOff, int H, int W, int Bn) {
  long long total = (long long)Bn * C * H * W;
  long long t = (long long)blockIdx.x * blockDim.x + threadIdx.x;
  if (t >= total) return;
  int w = (int)(t % W); long long r = t / W;
  int h = (int)(r % H); r /= H;
  int c = (int)(r % C); int b = (int)(r / C);
  Y[(((size_t)b * Ctot + chOff + c) * H + h) * W + w] = X[t];
}

__global__ void k_fill(float* __restrict__ p, long long n, float v) {
  for (long long i = (long long)blockIdx.x * blockDim.x + threadIdx.x; i < n;
       i += (long long)gridDim.x * blockDim.x)
    p[i] = v;
}

__global__ void k_minmax(const float* __restrict__ X, int stride, long long n,
                         unsigned* __restrict__ cell) {
  __shared__ unsigned smn[256], smx[256];
  unsigned mn = 0xFFFFFFFFu, mx = 0u;
  for (long long i = (long long)blockIdx.x * blockDim.x + threadIdx.x; i < n;
       i += (long long)gridDim.x * blockDim.x) {
    unsigned o = f2ord(X[i * (long long)stride]);
    mn = o < mn ? o : mn;
    mx = o > mx ? o : mx;
  }
  smn[threadIdx.x] = mn; smx[threadIdx.x] = mx;
  __syncthreads();
  for (int s = blockDim.x / 2; s > 0; s >>= 1) {
    if ((int)threadIdx.x < s) {
      unsigned a = smn[threadIdx.x + s];
      unsigned b = smx[threadIdx.x + s];
      if (a < smn[threadIdx.x]) smn[threadIdx.x] = a;
      if (b > smx[threadIdx.x]) smx[threadIdx.x] = b;
    }
    __syncthreads();
  }
  if (threadIdx.x == 0) { atomicMin(&cell[0], smn[0]); atomicMax(&cell[1], smx[0]); }
}

__global__ void k_mm_init(unsigned* __restrict__ cells, int npairs) {
  int t = blockIdx.x * blockDim.x + threadIdx.x;
  if (t < npairs) { cells[2 * t] = 0xFFFFFFFFu; cells[2 * t + 1] = 0u; }
}

__global__ void k_interp(const float* __restrict__ feats, const float* __restrict__ coo,
                         const unsigned* __restrict__ mm, float* __restrict__ fv,
                         int C, int Hc, int Wc, int N, int Bn) {
  int t = blockIdx.x * blockDim.x + threadIdx.x;
  int total = Bn * N;
  if (t >= total) return;
  int b = t / N;
  float y = coo[(size_t)t * 3 + 0];
  float x = coo[(size_t)t * 3 + 1];
  float z = coo[(size_t)t * 3 + 2];
  float ymin = ord2f(mm[0]), ymax = ord2f(mm[1]);
  float xmin = ord2f(mm[2]), xmax = ord2f(mm[3]);
  int fy = (int)((y - ymin) / (ymax - ymin) * (float)(Hc - 1));
  int fx = (int)((x - xmin) / (xmax - xmin) * (float)(Wc - 1));
  fy = fy < 0 ? 0 : (fy > Hc - 1 ? Hc - 1 : fy);
  fx = fx < 0 ? 0 : (fx > Wc - 1 ? Wc - 1 : fx);
  size_t idx = (size_t)fy * Wc + fx;
  const float* fb = feats + (size_t)b * C * Hc * Wc;
  float* out = fv + (size_t)t * (C + 3);
  for (int c = 0; c < C; ++c) out[c] = fb[(size_t)c * Hc * Wc + idx];
  out[C] = y; out[C + 1] = x; out[C + 2] = z;
}

__global__ void k_mask(const float* __restrict__ occ, const float* __restrict__ coo,
                       const float* __restrict__ rgb, float* __restrict__ ox,
                       float* __restrict__ oy, float* __restrict__ orgb, int total) {
  int t = blockIdx.x * blockDim.x + threadIdx.x;
  if (t >= total) return;
  float m = occ[t] > -1.0f ? 1.0f : 0.0f;
  ox[t] = coo[(size_t)t * 3 + 0] * m;
  oy[t] = coo[(size_t)t * 3 + 1] * m;
  orgb[(size_t)t * 3 + 0] = rgb[(size_t)t * 3 + 0] * m;
  orgb[(size_t)t * 3 + 1] = rgb[(size_t)t * 3 + 1] * m;
  orgb[(size_t)t * 3 + 2] = rgb[(size_t)t * 3 + 2] * m;
}

__global__ void k_scatter(const float* __restrict__ ox, const float* __restrict__ oy,
                          const float* __restrict__ orgb, const unsigned* __restrict__ mm,
                          float* __restrict__ img, int Hc, int Wc, int N, int Bn) {
  int t = blockIdx.x * blockDim.x + threadIdx.x;
  int total = Bn * N;
  if (t >= total) return;
  int b = t / N;
  float xmin = ord2f(mm[0]), xmax = ord2f(mm[1]);
  float ymin = ord2f(mm[2]), ymax = ord2f(mm[3]);
  int ix = (int)((ox[t] - xmin) / (xmax - xmin) * (float)(Wc - 1));
  int iy = (int)((oy[t] - ymin) / (ymax - ymin) * (float)(Hc - 1));
  ix = ix < 0 ? 0 : (ix > Wc - 1 ? Wc - 1 : ix);
  iy = iy < 0 ? 0 : (iy > Hc - 1 ? Hc - 1 : iy);
  for (int c = 0; c < 3; ++c)
    img[(((size_t)b * 3 + c) * Hc + iy) * Wc + ix] = orgb[(size_t)t * 3 + c] * 255.0f;
}

// =====================================================================
// Host orchestration
// =====================================================================
extern "C" void kernel_launch(void* const* d_in, const int* in_sizes, int n_in,
                              void* d_out, int out_size, void* d_ws, size_t ws_size,
                              hipStream_t stream) {
  (void)in_sizes; (void)n_in; (void)out_size; (void)ws_size;
  const int Bn = 2, N_VOX = 65536, HH = 256, WW = 256;

  const float* image = (const float*)d_in[0];
  const float* coo   = (const float*)d_in[1];
  auto prm = [&](int i) { return (const float*)d_in[i]; };
  // params flattened depth-first in setup_inputs() insertion order:
  //  geo (36 arrays) @2, ren (36) @38, occ mlp (8) @74, rgb mlp (8) @82

  float* ws = (float*)d_ws;
  size_t off = 0;
  auto fall = [&](size_t n) { float* r = ws + off; off += n; return r; };
  float* T1  = fall((size_t)Bn * 32 * 65536);
  float* D1  = fall((size_t)Bn * 32 * 65536);
  float* D2  = fall((size_t)Bn * 64 * 16384);
  float* D3  = fall((size_t)Bn * 128 * 4096);
  float* BNK = fall((size_t)Bn * 256 * 1024);
  float* PL  = fall((size_t)Bn * 32 * 16384);
  float* CT  = fall((size_t)Bn * 64 * 65536);
  float* UO  = fall((size_t)Bn * 32 * 65536);
  float* FE  = fall((size_t)Bn * 128 * 65536);
  float* FV  = fall((size_t)Bn * N_VOX * 131);
  float* H0  = fall((size_t)Bn * N_VOX * 131);
  float* H1  = fall((size_t)Bn * N_VOX * 131);
  unsigned* MM = (unsigned*)fall(64);
  float* OX   = fall((size_t)Bn * N_VOX);
  float* OY   = fall((size_t)Bn * N_VOX);
  float* ORGB = fall((size_t)Bn * N_VOX * 3);

  auto conv = [&](const float* X, int wi, int bi, float* Y, int Cin, int Cout,
                  int Hh, int Ww, int ks, int relu) {
    int P = Hh * Ww;
    dim3 g((P + 255) / 256, (Cout + 63) / 64, Bn);
    k_conv_wmma<<<g, 128, 0, stream>>>(X, prm(wi), prm(bi), Y, Cin, Cout, Hh, Ww,
                                       ks, ks == 3 ? 1 : 0, relu);
  };
  auto dc = [&](const float* X, int pb, float* Y, int Cin, int Cout, int Hh, int Ww) {
    conv(X, pb + 0, pb + 1, T1, Cin, Cout, Hh, Ww, 3, 1);
    conv(T1, pb + 2, pb + 3, Y, Cout, Cout, Hh, Ww, 3, 1);
  };
  auto pool = [&](const float* X, float* Y, int C, int Hh, int Ww) {
    long long total = (long long)Bn * C * (Hh / 2) * (Ww / 2);
    k_pool<<<(int)((total + 255) / 256), 256, 0, stream>>>(X, Y, C, Hh, Ww, Bn);
  };
  auto tconv = [&](const float* X, int wi, int bi, float* Y, int Cin, int Cout,
                   int Hin, int Win, int Ctot) {
    long long total = (long long)Bn * Cout * Hin * 2 * Win * 2;
    k_tconv<<<(int)((total + 255) / 256), 256, 0, stream>>>(X, prm(wi), prm(bi), Y,
                                                            Cin, Cout, Hin, Win, Ctot, Bn);
  };
  auto copych = [&](const float* X, float* Y, int C, int Ctot, int chOff, int Hh, int Ww) {
    long long total = (long long)Bn * C * Hh * Ww;
    k_copych<<<(int)((total + 255) / 256), 256, 0, stream>>>(X, Y, C, Ctot, chOff, Hh, Ww, Bn);
  };
  auto gemm = [&](const float* A, int wi, int bi, float* C, int M, int N, int K, int relu) {
    dim3 g((M + 63) / 64, (N + 63) / 64, 1);
    k_gemm_wmma<<<g, 128, 0, stream>>>(A, prm(wi), prm(bi), C, M, N, K, relu);
  };
  auto run_unet = [&](int pb, float* fe) {
    dc(image, pb + 0, D1, 3, 32, 256, 256);
    pool(D1, PL, 32, 256, 256);
    dc(PL, pb + 4, D2, 32, 64, 128, 128);
    pool(D2, PL, 64, 128, 128);
    dc(PL, pb + 8, D3, 64, 128, 64, 64);
    pool(D3, PL, 128, 64, 64);
    dc(PL, pb + 12, BNK, 128, 256, 32, 32);
    tconv(BNK, pb + 16, pb + 17, CT, 256, 128, 32, 32, 256);
    copych(D3, CT, 128, 256, 128, 64, 64);
    dc(CT, pb + 18, UO, 256, 128, 64, 64);
    tconv(UO, pb + 22, pb + 23, CT, 128, 64, 64, 64, 128);
    copych(D2, CT, 64, 128, 64, 128, 128);
    dc(CT, pb + 24, UO, 128, 64, 128, 128);
    tconv(UO, pb + 28, pb + 29, CT, 64, 32, 128, 128, 64);
    copych(D1, CT, 32, 64, 32, 256, 256);
    dc(CT, pb + 30, UO, 64, 32, 256, 256);
    conv(UO, pb + 34, pb + 35, fe, 32, 128, 256, 256, 1, 0);
  };
  auto interp = [&]() {
    long long ncoo = (long long)Bn * N_VOX;
    k_mm_init<<<1, 32, 0, stream>>>(MM, 2);
    k_minmax<<<256, 256, 0, stream>>>(coo + 0, 3, ncoo, MM + 0);  // y
    k_minmax<<<256, 256, 0, stream>>>(coo + 1, 3, ncoo, MM + 2);  // x
    k_interp<<<(Bn * N_VOX + 255) / 256, 256, 0, stream>>>(FE, coo, MM, FV, 128,
                                                           HH, WW, N_VOX, Bn);
  };

  const int M = Bn * N_VOX;  // 131072 MLP rows
  float* occ_out = (float*)d_out;            // [B,N,1]
  float* rgb_out = occ_out + (size_t)M;      // [B,N,3]
  float* img_out = rgb_out + (size_t)M * 3;  // [B,3,H,W]

  // ---- geometry branch: UNet -> interp -> occ MLP ----
  run_unet(2, FE);
  interp();
  gemm(FV, 74, 75, H0, M, 131, 131, 1);
  gemm(H0, 76, 77, H1, M, 131, 131, 1);
  gemm(H1, 78, 79, H0, M, 64, 131, 1);
  gemm(H0, 80, 81, occ_out, M, 1, 64, 0);

  // ---- render branch: UNet -> interp -> rgb MLP ----
  run_unet(38, FE);
  interp();
  gemm(FV, 82, 83, H0, M, 131, 131, 1);
  gemm(H0, 84, 85, H1, M, 131, 131, 1);
  gemm(H1, 86, 87, H0, M, 64, 131, 1);
  gemm(H0, 88, 89, rgb_out, M, 3, 64, 0);

  // ---- rasterize ----
  k_fill<<<1536, 256, 0, stream>>>(img_out, (long long)Bn * 3 * HH * WW, 0.0f);
  k_mask<<<(M + 255) / 256, 256, 0, stream>>>(occ_out, coo, rgb_out, OX, OY, ORGB, M);
  k_mm_init<<<1, 32, 0, stream>>>(MM, 2);
  k_minmax<<<256, 256, 0, stream>>>(OX, 1, (long long)M, MM + 0);  // x
  k_minmax<<<256, 256, 0, stream>>>(OY, 1, (long long)M, MM + 2);  // y
  k_scatter<<<(M + 255) / 256, 256, 0, stream>>>(OX, OY, ORGB, MM, img_out,
                                                 HH, WW, N_VOX, Bn);
}
